// PromptGenerator_19774029430986
// MI455X (gfx1250) — matrix operand order repaired and checked by reading, not compiled, not run on hardware
//
#include <hip/hip_runtime.h>
#include <hip/hip_bf16.h>

typedef __attribute__((ext_vector_type(16))) _Float16 v16h;
typedef __attribute__((ext_vector_type(8)))  float    v8f;
typedef __attribute__((ext_vector_type(4)))  float    f32x4;

#define EMB 128
#define HID 128
// K-total = 2*EMB = 256 -> 8 k-steps of 32; N = HID = 128 -> 8 n-tiles of 16.
#define NFRAG       64                      // 8 kb * 8 nt
#define LDS_PAD     48                      // bytes/lane/fragment (32B data + 16B pad)
#define FRAG_STRIDE (32 * LDS_PAD)          // 1536 B per fragment
#define LDS_B_BYTES (NFRAG * FRAG_STRIDE)   // 98304

// ---------------------------------------------------------------------------
// Prep: swizzle W1 [256][128] f32 into f16 WMMA B-fragments in workspace.
// Fragment (kb,nt), lane l (n=l&15, hi=l>>4), half t (0..15):
//   value = W1[kb*32 + (t<8 ? hi*8 + t : 16 + hi*8 + t-8)][nt*16 + n]
// ---------------------------------------------------------------------------
__global__ void prep_w1_frags(const float* __restrict__ W1,
                              _Float16* __restrict__ wsB) {
  int idx = blockIdx.x * blockDim.x + threadIdx.x;   // 0 .. 64*32*16-1
  if (idx >= NFRAG * 32 * 16) return;
  int t    = idx & 15;
  int lane = (idx >> 4) & 31;
  int frag = idx >> 9;
  int kb = frag >> 3, nt = frag & 7;
  int K   = kb * 32 + (t < 8 ? (lane >> 4) * 8 + t : 16 + (lane >> 4) * 8 + (t - 8));
  int col = nt * 16 + (lane & 15);
  wsB[idx] = (_Float16)W1[K * HID + col];
}

// --------------------------- inline-asm helpers ----------------------------
// Per-use B-fragment load: two ds_load_b128, counters managed explicitly so
// the compiler cannot hoist the (loop-invariant) fragments into 512 VGPRs.
__device__ __forceinline__ void lds_load2(unsigned addr, f32x4& lo, f32x4& hi) {
  asm volatile("ds_load_b128 %0, %2\n\t"
               "ds_load_b128 %1, %2 offset:16"
               : "=&v"(lo), "=&v"(hi) : "v"(addr) : "memory");
}
__device__ __forceinline__ void wait_dscnt_le2() {
  asm volatile("s_wait_dscnt 0x2" ::: "memory");
}
__device__ __forceinline__ void wait_dscnt_0() {
  asm volatile("s_wait_dscnt 0x0" ::: "memory");
}

// --------------------------- A-fragment helpers ----------------------------
struct A4 { f32x4 a, b, c, d; };
__device__ __forceinline__ A4 load_a(const float* row, int o) {
  A4 t;
  t.a = *(const f32x4*)(row + o);
  t.b = *(const f32x4*)(row + o + 4);
  t.c = *(const f32x4*)(row + o + 16);
  t.d = *(const f32x4*)(row + o + 20);
  return t;
}
__device__ __forceinline__ v16h cvt_a(A4 t) {
  v16h a;
  a[0]=(_Float16)t.a[0];  a[1]=(_Float16)t.a[1];  a[2]=(_Float16)t.a[2];  a[3]=(_Float16)t.a[3];
  a[4]=(_Float16)t.b[0];  a[5]=(_Float16)t.b[1];  a[6]=(_Float16)t.b[2];  a[7]=(_Float16)t.b[3];
  a[8]=(_Float16)t.c[0];  a[9]=(_Float16)t.c[1];  a[10]=(_Float16)t.c[2]; a[11]=(_Float16)t.c[3];
  a[12]=(_Float16)t.d[0]; a[13]=(_Float16)t.d[1]; a[14]=(_Float16)t.d[2]; a[15]=(_Float16)t.d[3];
  return a;
}

// ---------------------------------------------------------------------------
// Main: per wave, 16-edge M-tile x 128 N x 256 K via 64 v_wmma_f32_16x16x32_f16,
// B fragments streamed from LDS (depth-1 pipelined ds_load_b128 pairs),
// A fragments double-buffered; LayerNorm/ReLU/dot/sigmoid epilogue via shfl_xor.
// ---------------------------------------------------------------------------
__global__ __launch_bounds__(256)
void edge_mlp_kernel(const float* __restrict__ node_emb,
                     const int*   __restrict__ edge_index,
                     const _Float16* __restrict__ wsB,
                     const float* __restrict__ b1,
                     const float* __restrict__ ln_g,
                     const float* __restrict__ ln_b,
                     const float* __restrict__ W2,
                     const float* __restrict__ b2,
                     float* __restrict__ out,
                     int E) {
  extern __shared__ char smem[];   // B fragments: NFRAG * 32 lanes * 48B

  // Cooperative copy ws -> LDS (4096 16B chunks, 256 threads x 16), padded.
  {
    const f32x4* src = (const f32x4*)wsB;
    #pragma unroll
    for (int i = 0; i < 16; ++i) {
      int c = threadIdx.x + i * 256;
      int dstOff = (c >> 1) * LDS_PAD + (c & 1) * 16;
      *(f32x4*)(smem + dstOff) = src[c];
    }
  }
  __syncthreads();

  const int lane = threadIdx.x & 31;
  const int n    = lane & 15;
  const int hi   = lane >> 4;
  const unsigned bbase = (unsigned)(size_t)smem + (unsigned)(lane * LDS_PAD);

  // Per-lane column constants for its 8 columns (nt*16 + n).
  float b1c[8], gc[8], bc[8], w2c[8];
  #pragma unroll
  for (int nt = 0; nt < 8; ++nt) {
    int col = nt * 16 + n;
    b1c[nt] = b1[col]; gc[nt] = ln_g[col]; bc[nt] = ln_b[col]; w2c[nt] = W2[col];
  }
  const float bias2 = b2[0];

  const int wavesPerWG = blockDim.x >> 5;
  const int w  = blockIdx.x * wavesPerWG + (threadIdx.x >> 5);
  const int nw = gridDim.x * wavesPerWG;
  const int nBlocks = (E + 15) >> 4;

  for (int blk = w; blk < nBlocks; blk += nw) {
    const int base = blk * 16;
    int e = base + n; if (e >= E) e = E - 1;         // clamp tail gathers
    const int si = edge_index[e];
    const int di = edge_index[E + e];
    const float* srow = node_emb + (size_t)si * EMB;
    const float* drow = node_emb + (size_t)di * EMB;

    v8f c[8];
    #pragma unroll
    for (int nt = 0; nt < 8; ++nt) c[nt] = (v8f)0.0f;

    // ---- software-pipelined GEMM ----
    v16h aab[2];
    aab[0] = cvt_a(load_a(srow, hi * 8));            // A for kb=0

    f32x4 blo[2], bhi[2];
    lds_load2(bbase, blo[0], bhi[0]);                // B fragment 0 in flight

    #pragma unroll
    for (int kb = 0; kb < 8; ++kb) {
      A4 tn;
      if (kb < 7) {                                  // prefetch next A chunk
        int k2 = kb + 1;
        tn = load_a((k2 < 4) ? srow : drow, (k2 & 3) * 32 + hi * 8);
      }
      #pragma unroll
      for (int nt = 0; nt < 8; ++nt) {
        const int f   = kb * 8 + nt;
        const int cur = f & 1, nxt = cur ^ 1;
        const int pf  = (f < NFRAG - 1) ? (f + 1) : (NFRAG - 1);
        lds_load2(bbase + (unsigned)(pf * FRAG_STRIDE), blo[nxt], bhi[nxt]);
        wait_dscnt_le2();                            // fragment f resident
        v16h bf;
        ((f32x4*)&bf)[0] = blo[cur];
        ((f32x4*)&bf)[1] = bhi[cur];
        c[nt] = __builtin_amdgcn_wmma_f32_16x16x32_f16(
                    false, aab[kb & 1], false, bf, (short)0, c[nt], false, false);
      }
      if (kb < 7) aab[(kb + 1) & 1] = cvt_a(tn);     // write the *other* A buffer
    }
    wait_dscnt_0();                                  // drain before compiler ds ops

    // ---- epilogue: +b1, LayerNorm over 128 cols, ReLU, dot W2, sigmoid ----
    float s[8], s2[8];
    #pragma unroll
    for (int r = 0; r < 8; ++r) { s[r] = 0.0f; s2[r] = 0.0f; }
    #pragma unroll
    for (int nt = 0; nt < 8; ++nt) {
      #pragma unroll
      for (int r = 0; r < 8; ++r) {
        float x = c[nt][r] + b1c[nt];
        c[nt][r] = x;
        s[r] += x;
        s2[r] = fmaf(x, x, s2[r]);
      }
    }
    #pragma unroll
    for (int m = 1; m < 16; m <<= 1) {               // stays within 16-lane halves
      #pragma unroll
      for (int r = 0; r < 8; ++r) {
        s[r]  += __shfl_xor(s[r],  m);
        s2[r] += __shfl_xor(s2[r], m);
      }
    }
    float mean[8], rstd[8], wsum[8];
    #pragma unroll
    for (int r = 0; r < 8; ++r) {
      mean[r] = s[r] * (1.0f / 128.0f);
      float var = s2[r] * (1.0f / 128.0f) - mean[r] * mean[r];
      rstd[r] = rsqrtf(var + 1e-5f);
      wsum[r] = 0.0f;
    }
    #pragma unroll
    for (int nt = 0; nt < 8; ++nt) {
      #pragma unroll
      for (int r = 0; r < 8; ++r) {
        float y = (c[nt][r] - mean[r]) * rstd[r] * gc[nt] + bc[nt];
        y = fmaxf(y, 0.0f);
        wsum[r] = fmaf(y, w2c[nt], wsum[r]);
      }
    }
    #pragma unroll
    for (int m = 1; m < 16; m <<= 1) {
      #pragma unroll
      for (int r = 0; r < 8; ++r) wsum[r] += __shfl_xor(wsum[r], m);
    }
    if (n == 0) {                                    // lanes 0 and 16 store
      #pragma unroll
      for (int r = 0; r < 8; ++r) {
        int eo = base + hi * 8 + r;
        if (eo < E) {
          float x  = wsum[r] + bias2;
          float sg = 1.0f / (1.0f + __expf(-x));
          if (sg != sg) sg = 0.5f;                   // nan_to_num(nan=0.5)
          out[eo] = sg;
        }
      }
    }
  }
}

extern "C" void kernel_launch(void* const* d_in, const int* in_sizes, int n_in,
                              void* d_out, int out_size, void* d_ws, size_t ws_size,
                              hipStream_t stream) {
  const float* node_emb  = (const float*)d_in[0];
  const int*   edge_idx  = (const int*)d_in[1];
  const float* W1        = (const float*)d_in[2];
  const float* b1        = (const float*)d_in[3];
  const float* ln_g      = (const float*)d_in[4];
  const float* ln_b      = (const float*)d_in[5];
  const float* W2        = (const float*)d_in[6];
  const float* b2        = (const float*)d_in[7];
  float*       out       = (float*)d_out;
  _Float16*    wsB       = (_Float16*)d_ws;          // 64 KB of B fragments

  const int E = in_sizes[1] / 2;

  prep_w1_frags<<<128, 256, 0, stream>>>(W1, wsB);

  const int nBlocks = (E + 15) / 16;                 // 16-edge M-tiles
  int wgs = (nBlocks + 7) / 8;                       // 8 waves per WG
  if (wgs > 640) wgs = 640;
  if (wgs < 1)   wgs = 1;
  edge_mlp_kernel<<<wgs, 256, LDS_B_BYTES, stream>>>(
      node_emb, edge_idx, wsB, b1, ln_g, ln_b, W2, b2, out, E);
}
